// Grouped_Query_Attention_70454643523649
// MI455X (gfx1250) — matrix operand order repaired and checked by reading, not compiled, not run on hardware
//
#include <hip/hip_runtime.h>
#include <math.h>

typedef __bf16 bf16;
typedef __attribute__((ext_vector_type(16))) __bf16 v16bf;
typedef __attribute__((ext_vector_type(8)))  float  v8f;

// Problem constants (fixed by the reference).
constexpr int BB   = 2;
constexpr int NN   = 2048;
constexpr int DD   = 2048;   // D_IN == D_OUT
constexpr int HH   = 16;
constexpr int HDIM = 128;
constexpr int DKV  = 512;    // G * HD
constexpr int MROW = BB * NN;        // 4096
constexpr float SCALE = 0.08838834764831845f;  // 1/sqrt(128)
constexpr float BIGNEG = 1e30f;

// ---------------------------------------------------------------------------
// Async global->LDS staging (CDNA5 GLOBAL_LOAD_ASYNC_TO_LDS_B128, ASYNCcnt).
// The builtin takes int4* in addrspace(1)/(3) (per hipcc diagnostic), with
// immediate offset + cpol.  Fallback: synchronous uint4 copy.
// ---------------------------------------------------------------------------
#if defined(__has_builtin)
#if __has_builtin(__builtin_amdgcn_global_load_async_to_lds_b128) && \
    __has_builtin(__builtin_amdgcn_s_wait_asynccnt)
#define USE_ASYNC_LDS 1
#endif
#endif

#if defined(USE_ASYNC_LDS)
typedef int i32x4 __attribute__((ext_vector_type(4)));
typedef __attribute__((address_space(1))) i32x4 as1_i32x4;
typedef __attribute__((address_space(3))) i32x4 as3_i32x4;
#endif

__device__ __forceinline__ void stage16(const bf16* g, bf16* l) {
#if defined(USE_ASYNC_LDS)
  __builtin_amdgcn_global_load_async_to_lds_b128((as1_i32x4*)g, (as3_i32x4*)l,
                                                 0, 0);
#else
  *(uint4*)l = *(const uint4*)g;
#endif
}

__device__ __forceinline__ void stage_fence() {
#if defined(USE_ASYNC_LDS)
  __builtin_amdgcn_s_wait_asynccnt(0);
#endif
}

// ---------------------------------------------------------------------------
// WMMA fragment gathers from LDS (layouts per CDNA5 ISA 7.12.2, wave32).
// A 16x32 bf16: lane holds row M=lane&15; lanes<16 K={0..7,16..23},
//               lanes>=16 K={8..15,24..31}; 2 consecutive K per VGPR.
//               -> two contiguous 16B chunks = 2x ds_load_b128.
// B 32x16 bf16: lane holds col N=lane&15; lanes<16 K={0..7,16..23},
//               lanes>=16 K={8..15,24..31} (same K order as A).
// ---------------------------------------------------------------------------
__device__ __forceinline__ v16bf frag_a(const bf16* row, int half) {
  union { v16bf v; unsigned u[8]; } f;
#pragma unroll
  for (int i = 0; i < 4; ++i)
    f.u[i] = *(const unsigned*)(row + 8 * half + 2 * i);
#pragma unroll
  for (int i = 0; i < 4; ++i)
    f.u[4 + i] = *(const unsigned*)(row + 16 + 8 * half + 2 * i);
  return f.v;
}

__device__ __forceinline__ v16bf frag_b(const bf16* col, int half) {
  union { v16bf v; unsigned u[8]; } f;
#pragma unroll
  for (int i = 0; i < 4; ++i)
    f.u[i] = *(const unsigned*)(col + 8 * half + 2 * i);
#pragma unroll
  for (int i = 0; i < 4; ++i)
    f.u[4 + i] = *(const unsigned*)(col + 16 + 8 * half + 2 * i);
  return f.v;
}

__device__ __forceinline__ v8f wmma_bf16(v16bf a, v16bf b, v8f c) {
  return __builtin_amdgcn_wmma_f32_16x16x32_bf16(false, a, false, b, (short)0,
                                                 c, false, false);
}

// ---------------------------------------------------------------------------
// fp32 -> bf16 conversion (vectorized 4-wide).
// ---------------------------------------------------------------------------
__global__ __launch_bounds__(256) void cvt_f32_bf16(const float* __restrict__ s,
                                                    bf16* __restrict__ d, int n) {
  int i = (blockIdx.x * 256 + threadIdx.x) * 4;
  if (i >= n) return;
  float4 f = *(const float4*)(s + i);
  union { bf16 h[4]; uint2 u; } o;
  o.h[0] = (bf16)f.x; o.h[1] = (bf16)f.y; o.h[2] = (bf16)f.z; o.h[3] = (bf16)f.w;
  *(uint2*)(d + i) = o.u;
}

// ---------------------------------------------------------------------------
// Tiled bf16 GEMM: C[M,NC] = A[M,KK] @ B[KK,NC].  All strides constexpr.
// 256 threads (8 waves); 128x128 C tile; K-step 32.
// Wave tiling 4x2: wave w -> rows (w&3)*32, cols (w>>2)*64:
//   2 A-frags x 4 B-frags -> 8 WMMAs per 12 ds_load_b128.
// OUTMODE 0: bf16 output.  OUTMODE 1: f32 output + bias.
// ---------------------------------------------------------------------------
template <int NC, int KK, int OUTMODE>
__global__ __launch_bounds__(256) void gemm_bf16(
    const bf16* __restrict__ A, const bf16* __restrict__ Bm,
    bf16* __restrict__ Cb, float* __restrict__ Cf,
    const float* __restrict__ bias) {
  __shared__ bf16 sA[128][40];    // A tile row-major, padded (80B rows, 16B aligned)
  __shared__ bf16 sBt[128][40];   // B tile transposed: sBt[n][k]

  const int t    = threadIdx.x;
  const int w    = t >> 5;
  const int lane = t & 31;
  const int half = lane >> 4;
  const int ln   = lane & 15;
  const long m0 = (long)blockIdx.y * 128;
  const long n0 = (long)blockIdx.x * 128;
  const int r0 = (w & 3) * 32;    // wave row block
  const int c0 = (w >> 2) * 64;   // wave col block

  v8f acc[2][4] = {};

  const int rA = t >> 1, cA = (t & 1) * 16;
  const bf16* gA = A + (m0 + rA) * (long)KK + cA;
  const int pB = t >> 4, nB = (t & 15) * 8;
  const bf16* gB = Bm + (long)(2 * pB) * NC + n0 + nB;

  for (int k0 = 0; k0 < KK; k0 += 32) {
    {   // stage A tile: 128x32, 16 bf16 per thread (async -> LDS if available)
      stage16(gA + k0,     &sA[rA][cA]);
      stage16(gA + k0 + 8, &sA[rA][cA + 8]);
    }
    {   // stage B tile transposed: pack K-pairs -> 32-bit LDS words
      const bf16* g0 = gB + (long)k0 * NC;
      union { uint4 v; unsigned short s[8]; } q0, q1;
      q0.v = *(const uint4*)g0;
      q1.v = *(const uint4*)(g0 + NC);
#pragma unroll
      for (int e = 0; e < 8; ++e) {
        unsigned pk = (unsigned)q0.s[e] | ((unsigned)q1.s[e] << 16);
        *(unsigned*)&sBt[nB + e][2 * pB] = pk;
      }
    }
    if (k0 + 32 < KK) {             // pull next K-tile toward cache
      __builtin_prefetch(gA + k0 + 32, 0, 0);
      __builtin_prefetch(gB + (long)(k0 + 32) * NC, 0, 0);
    }
    stage_fence();
    __syncthreads();

    // batch all fragment loads, then run 8 WMMAs back-to-back
    v16bf af[2], bf_[4];
#pragma unroll
    for (int i = 0; i < 2; ++i) af[i] = frag_a(&sA[r0 + 16 * i + ln][0], half);
#pragma unroll
    for (int j = 0; j < 4; ++j) bf_[j] = frag_b(&sBt[c0 + 16 * j + ln][0], half);
#pragma unroll
    for (int i = 0; i < 2; ++i)
#pragma unroll
      for (int j = 0; j < 4; ++j)
        acc[i][j] = wmma_bf16(af[i], bf_[j], acc[i][j]);
    __syncthreads();
  }

#pragma unroll
  for (int i = 0; i < 2; ++i) {
    const long rowb = m0 + r0 + 16 * i + 8 * half;
#pragma unroll
    for (int j = 0; j < 4; ++j) {
      const long col = n0 + c0 + 16 * j + ln;
      if constexpr (OUTMODE == 0) {
        bf16* cp = Cb + rowb * NC + col;
#pragma unroll
        for (int v = 0; v < 8; ++v) cp[(long)v * NC] = (bf16)acc[i][j][v];
      } else {
        const float bv = bias[col];
        float* cp = Cf + rowb * NC + col;
#pragma unroll
        for (int v = 0; v < 8; ++v) cp[(long)v * NC] = acc[i][j][v] + bv;
      }
    }
  }
}

// ---------------------------------------------------------------------------
// Causal flash-attention, grouped KV.
// Grid: (N/64, H, B); 128 threads = 4 waves; wave w owns 16 q-rows.
// K processed in blocks of 32 keys; online softmax per row.
// ---------------------------------------------------------------------------
__global__ __launch_bounds__(128) void gqa_attn(
    const bf16* __restrict__ Q,   // [B,N,2048]  (h*128+d)
    const bf16* __restrict__ Kg,  // [B,N,512]   (g*128+d)
    const bf16* __restrict__ Vg,  // [B,N,512]
    bf16* __restrict__ ctx) {     // [B,N,2048]
  __shared__ bf16 Qs[64][136];    // q-rows x HD, padded (272B rows)
  __shared__ bf16 Ks[32][136];    // key tokens x HD (d-pairs contiguous)
  __shared__ bf16 Vt[128][40];    // transposed V: Vt[d][key] (key-pairs contiguous)
  __shared__ bf16 Ps[4][16][40];  // per-wave probabilities 16x32

  const int t    = threadIdx.x;
  const int w    = t >> 5;
  const int lane = t & 31;
  const int half = lane >> 4;
  const int ln   = lane & 15;
  const int qb = blockIdx.x, h = blockIdx.y, b = blockIdx.z;
  const int g  = h >> 2;          // REP = 4
  const int q0 = qb * 64;

  {   // stage Q block once: 64 x 128 (async -> LDS if available)
    int r = t >> 1, d0 = (t & 1) * 64;
    const bf16* gq = Q + ((long)(b * NN + q0 + r)) * DD + h * HDIM + d0;
#pragma unroll
    for (int i = 0; i < 8; ++i)
      stage16(gq + 8 * i, &Qs[r][d0 + 8 * i]);
  }

  float mi[8], li[8], alph[8];
  v8f o[8] = {};
#pragma unroll
  for (int v = 0; v < 8; ++v) { mi[v] = -BIGNEG; li[v] = 0.0f; }

  const int nkb = 2 * (qb + 1);   // causal: keys up to q0+63
  for (int kb = 0; kb < nkb; ++kb) {
    const int kk0 = kb * 32;
    {   // stage K block: 32 tokens x 128 (async -> LDS if available)
      int tok = t >> 2, d0 = (t & 3) * 32;
      const bf16* gk = Kg + ((long)(b * NN + kk0 + tok)) * DKV + g * HDIM + d0;
#pragma unroll
      for (int i = 0; i < 4; ++i)
        stage16(gk + 8 * i, &Ks[tok][d0 + 8 * i]);
    }
    {   // stage V block transposed (pack token pairs through registers)
      int p = t >> 3, d0 = (t & 7) * 16;
      const bf16* gv = Vg + ((long)(b * NN + kk0 + 2 * p)) * DKV + g * HDIM + d0;
      union { uint4 v; unsigned short s[8]; } a0, a1, b0, b1;
      a0.v = *(const uint4*)gv;          a1.v = *(const uint4*)(gv + 8);
      b0.v = *(const uint4*)(gv + DKV);  b1.v = *(const uint4*)(gv + DKV + 8);
#pragma unroll
      for (int e = 0; e < 8; ++e) {
        *(unsigned*)&Vt[d0 + e][2 * p]     = (unsigned)a0.s[e] | ((unsigned)b0.s[e] << 16);
        *(unsigned*)&Vt[d0 + 8 + e][2 * p] = (unsigned)a1.s[e] | ((unsigned)b1.s[e] << 16);
      }
    }
    stage_fence();
    __syncthreads();

    // S = Q(16x128) @ K^T(128x32): Q A-frags are ksub-invariant -> hoist
    v16bf aq[4];
#pragma unroll
    for (int c = 0; c < 4; ++c) aq[c] = frag_a(&Qs[16 * w + ln][32 * c], half);

    v8f s[2] = {};
#pragma unroll
    for (int ks = 0; ks < 2; ++ks) {
      v16bf bk[4];
#pragma unroll
      for (int c = 0; c < 4; ++c) bk[c] = frag_b(&Ks[16 * ks + ln][32 * c], half);
#pragma unroll
      for (int c = 0; c < 4; ++c) s[ks] = wmma_bf16(aq[c], bk[c], s[ks]);
    }

    // mask + online softmax (rows live in 16-lane halves of the wave)
#pragma unroll
    for (int v = 0; v < 8; ++v) {
      float s0 = s[0][v] * SCALE, s1 = s[1][v] * SCALE;
      int q = q0 + 16 * w + v + 8 * half;
      if (kk0 + ln      > q) s0 = -BIGNEG;
      if (kk0 + 16 + ln > q) s1 = -BIGNEG;
      float rmax = fmaxf(s0, s1);
#pragma unroll
      for (int m = 1; m < 16; m <<= 1) rmax = fmaxf(rmax, __shfl_xor(rmax, m, 32));
      float mnew = fmaxf(mi[v], rmax);
      float a  = __expf(mi[v] - mnew);
      float p0 = __expf(s0 - mnew);
      float p1 = __expf(s1 - mnew);
      float rs = p0 + p1;
#pragma unroll
      for (int m = 1; m < 16; m <<= 1) rs += __shfl_xor(rs, m, 32);
      li[v] = li[v] * a + rs;
      mi[v] = mnew;
      alph[v] = a;
      int rl = v + 8 * half;
      Ps[w][rl][ln]      = (bf16)p0;   // re-layout C-frag -> A-frag via LDS
      Ps[w][rl][16 + ln] = (bf16)p1;
    }
#pragma unroll
    for (int j = 0; j < 8; ++j)
#pragma unroll
      for (int v = 0; v < 8; ++v) o[j][v] *= alph[v];

    // O += P(16x32) @ V(32x128): batch V B-frags, then 8 WMMAs
    v16bf ap = frag_a(&Ps[w][ln][0], half);
    v16bf bv[8];
#pragma unroll
    for (int j = 0; j < 8; ++j) bv[j] = frag_b(&Vt[16 * j + ln][0], half);
#pragma unroll
    for (int j = 0; j < 8; ++j) o[j] = wmma_bf16(ap, bv[j], o[j]);
    __syncthreads();
  }

  // normalize and write context (bf16) at [b, q, h*128 + d]
#pragma unroll
  for (int v = 0; v < 8; ++v) {
    float inv = 1.0f / li[v];
    int q = q0 + 16 * w + v + 8 * half;
    bf16* gc = ctx + ((long)(b * NN + q)) * DD + h * HDIM + ln;
#pragma unroll
    for (int j = 0; j < 8; ++j) gc[16 * j] = (bf16)(o[j][v] * inv);
  }
}

// ---------------------------------------------------------------------------
// Host-side orchestration.
// ---------------------------------------------------------------------------
extern "C" void kernel_launch(void* const* d_in, const int* in_sizes, int n_in,
                              void* d_out, int out_size, void* d_ws, size_t ws_size,
                              hipStream_t stream) {
  const float* x  = (const float*)d_in[0];
  const float* Wq = (const float*)d_in[1];
  const float* Wk = (const float*)d_in[2];
  const float* Wv = (const float*)d_in[3];
  const float* Wo = (const float*)d_in[4];
  const float* bo = (const float*)d_in[5];
  float* out = (float*)d_out;

  const long nX  = (long)MROW * DD;   // 8,388,608
  const long nWq = (long)DD * DD;     // 4,194,304
  const long nWk = (long)DD * DKV;    // 1,048,576
  const long nK  = (long)MROW * DKV;  // 2,097,152

  bf16* p    = (bf16*)d_ws;
  bf16* xb   = p;             p += nX;
  bf16* Wqb  = p;             p += nWq;
  bf16* Wkb  = p;             p += nWk;
  bf16* Wvb  = p;             p += nWk;
  bf16* Wob  = p;             p += nWq;
  bf16* Qb   = p;             p += nX;
  bf16* Kb   = p;             p += nK;
  bf16* Vb   = p;             p += nK;
  bf16* Ctxb = p;             p += nX;

  auto cvt = [&](const float* s, bf16* d, long n) {
    cvt_f32_bf16<<<dim3((unsigned)((n / 4 + 255) / 256)), 256, 0, stream>>>(s, d, (int)n);
  };
  cvt(x,  xb,  nX);
  cvt(Wq, Wqb, nWq);
  cvt(Wk, Wkb, nWk);
  cvt(Wv, Wvb, nWk);
  cvt(Wo, Wob, nWq);

  // Projections: Q [4096x2048], K/V [4096x512]
  gemm_bf16<DD, DD, 0><<<dim3(DD / 128, MROW / 128), 256, 0, stream>>>(
      xb, Wqb, Qb, nullptr, nullptr);
  gemm_bf16<DKV, DD, 0><<<dim3(DKV / 128, MROW / 128), 256, 0, stream>>>(
      xb, Wkb, Kb, nullptr, nullptr);
  gemm_bf16<DKV, DD, 0><<<dim3(DKV / 128, MROW / 128), 256, 0, stream>>>(
      xb, Wvb, Vb, nullptr, nullptr);

  // Attention: (N/64, H, B)
  gqa_attn<<<dim3(NN / 64, HH, BB), 128, 0, stream>>>(Qb, Kb, Vb, Ctxb);

  // Output projection (+bias) -> fp32 d_out
  gemm_bf16<DD, DD, 1><<<dim3(DD / 128, MROW / 128), 256, 0, stream>>>(
      Ctxb, Wob, nullptr, out, bo);
}
